// TheExpertAbundance_69449621176824
// MI455X (gfx1250) — matrix operand order repaired and checked by reading, not compiled, not run on hardware
//
#include <hip/hip_runtime.h>
#include <hip/hip_bf16.h>
#include <math.h>

// Problem constants (match reference)
#define BB 4
#define TT 2048
#define EE 2048
#define HH 16
#define DD 128

typedef __attribute__((ext_vector_type(16))) __bf16 bf16x16;
typedef __attribute__((ext_vector_type(8)))  float  v8f;

// CDNA5 async global->LDS path (ASYNCcnt).  The builtin takes
// (int4 addrspace(1)* gsrc, int4 addrspace(3)* ldst, imm offset, imm cpol).
#if defined(__has_builtin)
#if __has_builtin(__builtin_amdgcn_global_load_async_to_lds_b128)
#define HAVE_ASYNC_LDS 1
#endif
#endif

#ifdef HAVE_ASYNC_LDS
typedef int v4i __attribute__((vector_size(16)));
typedef __attribute__((address_space(1))) v4i* gas_ptr;
typedef __attribute__((address_space(3))) v4i* las_ptr;
#define ASYNC_CP16(g, l)                                                        \
    __builtin_amdgcn_global_load_async_to_lds_b128((gas_ptr)(g), (las_ptr)(l), 0, 0)
#define ASYNC_FENCE() asm volatile("s_wait_asynccnt 0x0" ::: "memory")
#else
#define ASYNC_FENCE()
#endif

__device__ __forceinline__ unsigned short f2bf(float f) {
    union { __bf16 b; unsigned short u; } c; c.b = (__bf16)f; return c.u;
}

// Build a 16-element bf16 fragment from two 16-byte chunks.
__device__ __forceinline__ bf16x16 load_frag(const unsigned short* p0, const unsigned short* p1) {
    union { bf16x16 v; uint4 u[2]; } f;
    f.u[0] = *(const uint4*)p0;
    f.u[1] = *(const uint4*)p1;
    return f.v;
}

__device__ __forceinline__ v8f wmma_bf16(bf16x16 a, bf16x16 b, v8f c) {
    return __builtin_amdgcn_wmma_f32_16x16x32_bf16(false, a, false, b, (short)0, c, false, false);
}

// ---------------------------------------------------------------------------
// K0: f32 -> bf16 weight conversion (one pass; weights then live in L2)
// ---------------------------------------------------------------------------
__global__ __launch_bounds__(256) void cvt_f32_bf16(const float* __restrict__ src,
                                                    unsigned short* __restrict__ dst, int n) {
    int i = (blockIdx.x * 256 + threadIdx.x) * 4;
    if (i < n) {
        float4 v = *(const float4*)(src + i);
        union { unsigned short h[4]; uint2 u; } o;
        o.h[0] = f2bf(v.x); o.h[1] = f2bf(v.y); o.h[2] = f2bf(v.z); o.h[3] = f2bf(v.w);
        *(uint2*)(dst + i) = o.u;
    }
}

// ---------------------------------------------------------------------------
// K1: RMSNorm(x) -> bf16.  One block per (b,t) row of E=2048.
// ---------------------------------------------------------------------------
__global__ __launch_bounds__(256) void rmsnorm_cast(const float* __restrict__ x,
                                                    unsigned short* __restrict__ xn) {
    int row = blockIdx.x, tid = threadIdx.x;
    const float* xr = x + (size_t)row * EE;
    float4 a = *(const float4*)(xr + tid * 8);
    float4 b = *(const float4*)(xr + tid * 8 + 4);
    float ss = a.x*a.x + a.y*a.y + a.z*a.z + a.w*a.w
             + b.x*b.x + b.y*b.y + b.z*b.z + b.w*b.w;
    for (int off = 16; off; off >>= 1) ss += __shfl_xor(ss, off, 32);
    __shared__ float red[8];
    int wave = tid >> 5, lane = tid & 31;
    if (lane == 0) red[wave] = ss;
    __syncthreads();
    float tot = 0.f;
    #pragma unroll
    for (int i = 0; i < 8; i++) tot += red[i];
    float rstd = rsqrtf(tot * (1.0f / EE) + 1.1920929e-07f);
    union { unsigned short h[8]; uint4 u; } o;
    o.h[0]=f2bf(a.x*rstd); o.h[1]=f2bf(a.y*rstd); o.h[2]=f2bf(a.z*rstd); o.h[3]=f2bf(a.w*rstd);
    o.h[4]=f2bf(b.x*rstd); o.h[5]=f2bf(b.y*rstd); o.h[6]=f2bf(b.z*rstd); o.h[7]=f2bf(b.w*rstd);
    *(uint4*)(xn + (size_t)row * EE + tid * 8) = o.u;
}

// ---------------------------------------------------------------------------
// K2: QKV GEMM (bf16 WMMA, double-buffered async LDS) + fused RoPE + RMS.
//   grid = (M/128, 3H).  8 waves/WG, wave tile = 16M x 128N.
// ---------------------------------------------------------------------------
__global__ __launch_bounds__(256) void qkv_rope_kernel(
    const unsigned short* __restrict__ xn, const unsigned short* __restrict__ wq,
    const float* __restrict__ cosb, const float* __restrict__ sinb,
    unsigned short* __restrict__ qo, unsigned short* __restrict__ ko,
    unsigned short* __restrict__ vT) {
    __shared__ __align__(16) unsigned short ldsA[2 * 128 * 40];
    __shared__ __align__(16) unsigned short ldsB[2 * 128 * 40];
    __shared__ __align__(16) unsigned short tbuf[8][16 * 24];
    int tid = threadIdx.x, wave = tid >> 5, lane = tid & 31;
    int m0 = blockIdx.x * 128;
    int by = blockIdx.y;
    int kind = by >> 4, h = by & 15;

    // staging role: each thread owns one 32B chunk per tile
    int srow = tid >> 1, shf = tid & 1;
    const unsigned short* Ap = xn + (size_t)(m0 + srow) * EE + shf * 16;
    const unsigned short* Bp = wq + ((size_t)by * 128 + srow) * EE + shf * 16;
    unsigned short* sa = ldsA + srow * 40 + shf * 16;
    unsigned short* sb = ldsB + srow * 40 + shf * 16;

    v8f acc[8] = {};
#ifdef HAVE_ASYNC_LDS
    ASYNC_CP16(Ap, sa); ASYNC_CP16(Ap + 8, sa + 8);
    ASYNC_CP16(Bp, sb); ASYNC_CP16(Bp + 8, sb + 8);
#else
    {
        uint4 ra0 = *(const uint4*)Ap, ra1 = *(const uint4*)(Ap + 8);
        uint4 rb0 = *(const uint4*)Bp, rb1 = *(const uint4*)(Bp + 8);
        *(uint4*)sa = ra0; *(uint4*)(sa + 8) = ra1;
        *(uint4*)sb = rb0; *(uint4*)(sb + 8) = rb1;
    }
#endif
    int cur = 0;
    for (int k0 = 0; k0 < EE; k0 += 32) {
        ASYNC_FENCE();
        __syncthreads();
        int nxt = cur ^ 1;
        bool more = (k0 + 32 < EE);
#ifdef HAVE_ASYNC_LDS
        if (more) {
            ASYNC_CP16(Ap + k0 + 32, sa + nxt * 5120);
            ASYNC_CP16(Ap + k0 + 40, sa + nxt * 5120 + 8);
            ASYNC_CP16(Bp + k0 + 32, sb + nxt * 5120);
            ASYNC_CP16(Bp + k0 + 40, sb + nxt * 5120 + 8);
        }
#else
        uint4 na0 = {}, na1 = {}, nb0 = {}, nb1 = {};
        if (more) {
            na0 = *(const uint4*)(Ap + k0 + 32); na1 = *(const uint4*)(Ap + k0 + 40);
            nb0 = *(const uint4*)(Bp + k0 + 32); nb1 = *(const uint4*)(Bp + k0 + 40);
        }
#endif
        // compute from buffer `cur`; all B frags hoisted for staged waits
        const unsigned short* ar =
            ldsA + cur * 5120 + (wave * 16 + (lane & 15)) * 40 + ((lane >> 4) << 4);
        bf16x16 af = load_frag(ar, ar + 8);
        const unsigned short* bbase =
            ldsB + cur * 5120 + (lane & 15) * 40 + ((lane >> 4) << 4);
        bf16x16 bfr[8];
        #pragma unroll
        for (int nt = 0; nt < 8; nt++)
            bfr[nt] = load_frag(bbase + nt * 640, bbase + nt * 640 + 8);
        #pragma unroll
        for (int nt = 0; nt < 8; nt++)
            acc[nt] = wmma_bf16(af, bfr[nt], acc[nt]);
#ifndef HAVE_ASYNC_LDS
        if (more) {
            unsigned short* da = sa + nxt * 5120;
            unsigned short* db = sb + nxt * 5120;
            *(uint4*)da = na0; *(uint4*)(da + 8) = na1;
            *(uint4*)db = nb0; *(uint4*)(db + 8) = nb1;
        }
#endif
        cur = nxt;
    }

    int b = m0 >> 11;                    // m0 / T
    int tbase = (m0 & (TT - 1)) + wave * 16;
    int bh = b * HH + h;

    if (kind < 2) {
        // RoPE: col d (tile nt) pairs with col d+64 (tile nt+4), same lane.
        #pragma unroll
        for (int nt = 0; nt < 4; nt++) {
            int jj = nt * 16 + (lane & 15);       // 0..63
            #pragma unroll
            for (int r = 0; r < 8; r++) {
                int trow = tbase + ((lane < 16) ? r : 8 + r);
                float c = cosb[trow * 64 + jj];
                float s = sinb[trow * 64 + jj];
                float x1 = acc[nt][r], x2 = acc[nt + 4][r];
                acc[nt][r]     = x1 * c + x2 * s;
                acc[nt + 4][r] = x2 * c - x1 * s;
            }
        }
        // RMS over D=128 (per row), fold 1/sqrt(D) into q
        float scale = (kind == 0) ? 0.08838834764831845f : 1.0f;
        #pragma unroll
        for (int r = 0; r < 8; r++) {
            float ss = 0.f;
            #pragma unroll
            for (int nt = 0; nt < 8; nt++) ss += acc[nt][r] * acc[nt][r];
            ss += __shfl_xor(ss, 1, 32); ss += __shfl_xor(ss, 2, 32);
            ss += __shfl_xor(ss, 4, 32); ss += __shfl_xor(ss, 8, 32);
            float rstd = rsqrtf(ss * (1.0f / DD) + 1.1920929e-07f) * scale;
            #pragma unroll
            for (int nt = 0; nt < 8; nt++) acc[nt][r] *= rstd;
        }
        unsigned short* dstbase = ((kind == 0) ? qo : ko) + ((size_t)bh * TT + tbase) * DD;
        unsigned short* tb = tbuf[wave];
        #pragma unroll
        for (int nt = 0; nt < 8; nt++) {
            #pragma unroll
            for (int r = 0; r < 8; r++) {
                int row = (lane < 16) ? r : 8 + r;
                tb[row * 24 + (lane & 15)] = f2bf(acc[nt][r]);
            }
            asm volatile("s_wait_dscnt 0x0" ::: "memory");   // same-wave LDS in-order
            int row = lane >> 1, part = lane & 1;
            uint4 d = *(const uint4*)(tb + row * 24 + part * 8);
            *(uint4*)(dstbase + (size_t)row * DD + nt * 16 + part * 8) = d;
        }
    } else {
        // v: write transposed [b,h,d,t]; each lane's 8 C values are 8
        // consecutive t's at one d -> single 16B store.
        #pragma unroll
        for (int nt = 0; nt < 8; nt++) {
            union { unsigned short h8[8]; uint4 u; } o;
            #pragma unroll
            for (int r = 0; r < 8; r++) o.h8[r] = f2bf(acc[nt][r]);
            unsigned short* d = vT + ((size_t)bh * DD + nt * 16 + (lane & 15)) * TT
                                   + tbase + ((lane < 16) ? 0 : 8);
            *(uint4*)d = o.u;
        }
    }
}

// ---------------------------------------------------------------------------
// K3: Flash attention.  grid=(T/128, B*H); wave owns 16 q rows, online softmax.
// ---------------------------------------------------------------------------
__global__ __launch_bounds__(256) void attn_kernel(
    const unsigned short* __restrict__ q, const unsigned short* __restrict__ k,
    const unsigned short* __restrict__ vT, unsigned short* __restrict__ y) {
    __shared__ __align__(16) unsigned short pbuf[8][16 * 136];
    int tid = threadIdx.x, wave = tid >> 5, lane = tid & 31;
    int qi = blockIdx.x, bh = blockIdx.y;
    int b = bh >> 4, h = bh & 15;
    int q0 = qi * 128;
    const unsigned short* qb = q  + (size_t)bh * TT * DD;
    const unsigned short* kp = k  + (size_t)bh * TT * DD;
    const unsigned short* vb = vT + (size_t)bh * DD * TT;

    // Q fragments (K=D=128 -> 4 frags), resident for the whole loop
    bf16x16 qf[4];
    {
        int trow = q0 + wave * 16 + (lane & 15);
        const unsigned short* rp = qb + (size_t)trow * DD + ((lane >> 4) << 4);
        #pragma unroll
        for (int ds = 0; ds < 4; ds++) qf[ds] = load_frag(rp + ds * 32, rp + ds * 32 + 8);
    }

    v8f acc_o[8] = {};
    float mrow[8], lrow[8];
    #pragma unroll
    for (int r = 0; r < 8; r++) { mrow[r] = -3.0e38f; lrow[r] = 0.f; }
    unsigned short* pb = pbuf[wave];

    for (int kb = 0; kb <= qi; kb++) {
        if (kb < qi) {  // warm L2/L0 for next k/v tiles
            __builtin_prefetch(kp + (size_t)((kb + 1) * 128 + lane) * DD, 0, 3);
            __builtin_prefetch(vb + (size_t)lane * TT + (kb + 1) * 128, 0, 3);
        }
        // S = (q/sqrt(D)) . K^T  -- process nt in pairs, frags hoisted
        v8f s[8] = {};
        #pragma unroll
        for (int np = 0; np < 8; np += 2) {
            const unsigned short* c0 =
                kp + (size_t)(kb * 128 + np * 16 + (lane & 15)) * DD + ((lane >> 4) << 4);
            const unsigned short* c1 = c0 + (size_t)16 * DD;
            bf16x16 kf0[4], kf1[4];
            #pragma unroll
            for (int ds = 0; ds < 4; ds++) {
                kf0[ds] = load_frag(c0 + ds * 32, c0 + ds * 32 + 8);
                kf1[ds] = load_frag(c1 + ds * 32, c1 + ds * 32 + 8);
            }
            #pragma unroll
            for (int ds = 0; ds < 4; ds++) {
                s[np]     = wmma_bf16(qf[ds], kf0[ds], s[np]);
                s[np + 1] = wmma_bf16(qf[ds], kf1[ds], s[np + 1]);
            }
        }
        if (kb == qi) {  // causal mask on diagonal block
            #pragma unroll
            for (int nt = 0; nt < 8; nt++) {
                int kcol = kb * 128 + nt * 16 + (lane & 15);
                #pragma unroll
                for (int r = 0; r < 8; r++) {
                    int qrow = q0 + wave * 16 + ((lane < 16) ? r : 8 + r);
                    if (kcol > qrow) s[nt][r] = -3.0e38f;
                }
            }
        }
        // online softmax (row stats via 16-lane shuffle reductions)
        #pragma unroll
        for (int r = 0; r < 8; r++) {
            float m = s[0][r];
            #pragma unroll
            for (int nt = 1; nt < 8; nt++) m = fmaxf(m, s[nt][r]);
            m = fmaxf(m, __shfl_xor(m, 1, 32)); m = fmaxf(m, __shfl_xor(m, 2, 32));
            m = fmaxf(m, __shfl_xor(m, 4, 32)); m = fmaxf(m, __shfl_xor(m, 8, 32));
            float mnew = fmaxf(mrow[r], m);
            float alpha = __expf(mrow[r] - mnew);
            mrow[r] = mnew;
            float rs = 0.f;
            #pragma unroll
            for (int nt = 0; nt < 8; nt++) {
                float p = __expf(s[nt][r] - mnew);
                s[nt][r] = p; rs += p;
            }
            rs += __shfl_xor(rs, 1, 32); rs += __shfl_xor(rs, 2, 32);
            rs += __shfl_xor(rs, 4, 32); rs += __shfl_xor(rs, 8, 32);
            lrow[r] = lrow[r] * alpha + rs;
            #pragma unroll
            for (int nt = 0; nt < 8; nt++) acc_o[nt][r] *= alpha;
            int row = (lane < 16) ? r : 8 + r;   // P: C-layout -> LDS row-major
            #pragma unroll
            for (int nt = 0; nt < 8; nt++)
                pb[row * 136 + nt * 16 + (lane & 15)] = f2bf(s[nt][r]);
        }
        asm volatile("s_wait_dscnt 0x0" ::: "memory");
        // O += P @ V  (P as A-frags from LDS, V frags contiguous from vT)
        bf16x16 pf[4];
        {
            const unsigned short* rp = pb + (lane & 15) * 136 + ((lane >> 4) << 4);
            #pragma unroll
            for (int kt = 0; kt < 4; kt++) pf[kt] = load_frag(rp + kt * 32, rp + kt * 32 + 8);
        }
        #pragma unroll
        for (int np = 0; np < 8; np += 2) {
            const unsigned short* c0 =
                vb + (size_t)(np * 16 + (lane & 15)) * TT + kb * 128 + ((lane >> 4) << 4);
            const unsigned short* c1 = c0 + (size_t)16 * TT;
            bf16x16 vf0[4], vf1[4];
            #pragma unroll
            for (int kt = 0; kt < 4; kt++) {
                vf0[kt] = load_frag(c0 + kt * 32, c0 + kt * 32 + 8);
                vf1[kt] = load_frag(c1 + kt * 32, c1 + kt * 32 + 8);
            }
            #pragma unroll
            for (int kt = 0; kt < 4; kt++) {
                acc_o[np]     = wmma_bf16(pf[kt], vf0[kt], acc_o[np]);
                acc_o[np + 1] = wmma_bf16(pf[kt], vf1[kt], acc_o[np + 1]);
            }
        }
    }
    // normalize + write y[b, t, h*D + d] (bf16) through per-wave LDS transpose
    #pragma unroll
    for (int r = 0; r < 8; r++) {
        float inv = 1.0f / lrow[r];
        int row = (lane < 16) ? r : 8 + r;
        #pragma unroll
        for (int nt = 0; nt < 8; nt++)
            pb[row * 136 + nt * 16 + (lane & 15)] = f2bf(acc_o[nt][r] * inv);
    }
    asm volatile("s_wait_dscnt 0x0" ::: "memory");
    {
        int row = lane >> 1, part = lane & 1;
        const unsigned short* src = pb + row * 136 + part * 64;
        unsigned short* dst = y + ((size_t)b * TT + q0 + wave * 16 + row) * EE + h * DD + part * 64;
        #pragma unroll
        for (int c = 0; c < 4; c++) ((uint4*)dst)[c] = ((const uint4*)src)[c];
    }
}

// ---------------------------------------------------------------------------
// K4: SwiGLU GEMM (double-buffered).  WG computes matching u-block and w-block
//     (offset E) so u*silu(w) is register-elementwise.  grid=(M/128, E/64).
// ---------------------------------------------------------------------------
__global__ __launch_bounds__(256) void swiglu_kernel(
    const unsigned short* __restrict__ y, const unsigned short* __restrict__ wsw,
    unsigned short* __restrict__ hb) {
    __shared__ __align__(16) unsigned short ldsA[2 * 128 * 40];
    __shared__ __align__(16) unsigned short ldsU[2 * 64 * 40];
    __shared__ __align__(16) unsigned short ldsW[2 * 64 * 40];
    __shared__ __align__(16) unsigned short tbuf[8][16 * 24];
    int tid = threadIdx.x, wave = tid >> 5, lane = tid & 31;
    int m0 = blockIdx.x * 128, g0 = blockIdx.y * 64;

    int arow = tid >> 1, ahf = tid & 1;
    const unsigned short* Ap = y + (size_t)(m0 + arow) * EE + ahf * 16;
    unsigned short* sa = ldsA + arow * 40 + ahf * 16;
    int brow = tid >> 2, bq = tid & 3;
    const unsigned short* Up = wsw + (size_t)(g0 + brow) * EE + bq * 8;
    const unsigned short* Wp = wsw + (size_t)(EE + g0 + brow) * EE + bq * 8;
    unsigned short* su = ldsU + brow * 40 + bq * 8;
    unsigned short* sw = ldsW + brow * 40 + bq * 8;

    v8f au[4] = {}, aw[4] = {};
#ifdef HAVE_ASYNC_LDS
    ASYNC_CP16(Ap, sa); ASYNC_CP16(Ap + 8, sa + 8);
    ASYNC_CP16(Up, su); ASYNC_CP16(Wp, sw);
#else
    {
        uint4 ra0 = *(const uint4*)Ap, ra1 = *(const uint4*)(Ap + 8);
        uint4 ru = *(const uint4*)Up, rw = *(const uint4*)Wp;
        *(uint4*)sa = ra0; *(uint4*)(sa + 8) = ra1;
        *(uint4*)su = ru; *(uint4*)sw = rw;
    }
#endif
    int cur = 0;
    for (int k0 = 0; k0 < EE; k0 += 32) {
        ASYNC_FENCE();
        __syncthreads();
        int nxt = cur ^ 1;
        bool more = (k0 + 32 < EE);
#ifdef HAVE_ASYNC_LDS
        if (more) {
            ASYNC_CP16(Ap + k0 + 32, sa + nxt * 5120);
            ASYNC_CP16(Ap + k0 + 40, sa + nxt * 5120 + 8);
            ASYNC_CP16(Up + k0 + 32, su + nxt * 2560);
            ASYNC_CP16(Wp + k0 + 32, sw + nxt * 2560);
        }
#else
        uint4 na0 = {}, na1 = {}, nu = {}, nw = {};
        if (more) {
            na0 = *(const uint4*)(Ap + k0 + 32); na1 = *(const uint4*)(Ap + k0 + 40);
            nu = *(const uint4*)(Up + k0 + 32);  nw = *(const uint4*)(Wp + k0 + 32);
        }
#endif
        const unsigned short* ar =
            ldsA + cur * 5120 + (wave * 16 + (lane & 15)) * 40 + ((lane >> 4) << 4);
        bf16x16 af = load_frag(ar, ar + 8);
        const unsigned short* ub = ldsU + cur * 2560 + (lane & 15) * 40 + ((lane >> 4) << 4);
        const unsigned short* wb = ldsW + cur * 2560 + (lane & 15) * 40 + ((lane >> 4) << 4);
        bf16x16 uf[4], wf[4];
        #pragma unroll
        for (int nt = 0; nt < 4; nt++) {
            uf[nt] = load_frag(ub + nt * 640, ub + nt * 640 + 8);
            wf[nt] = load_frag(wb + nt * 640, wb + nt * 640 + 8);
        }
        #pragma unroll
        for (int nt = 0; nt < 4; nt++) {
            au[nt] = wmma_bf16(af, uf[nt], au[nt]);
            aw[nt] = wmma_bf16(af, wf[nt], aw[nt]);
        }
#ifndef HAVE_ASYNC_LDS
        if (more) {
            unsigned short* da = sa + nxt * 5120;
            *(uint4*)da = na0; *(uint4*)(da + 8) = na1;
            *(uint4*)(su + nxt * 2560) = nu;
            *(uint4*)(sw + nxt * 2560) = nw;
        }
#endif
        cur = nxt;
    }
    unsigned short* tb = tbuf[wave];
    unsigned short* dstbase = hb + (size_t)(m0 + wave * 16) * EE + g0;
    #pragma unroll
    for (int nt = 0; nt < 4; nt++) {
        #pragma unroll
        for (int r = 0; r < 8; r++) {
            float u = au[nt][r], w = aw[nt][r];
            float hval = u * w / (1.0f + __expf(-w));    // u * silu(w)
            int row = (lane < 16) ? r : 8 + r;
            tb[row * 24 + (lane & 15)] = f2bf(hval);
        }
        asm volatile("s_wait_dscnt 0x0" ::: "memory");
        int row = lane >> 1, part = lane & 1;
        uint4 d = *(const uint4*)(tb + row * 24 + part * 8);
        *(uint4*)(dstbase + (size_t)row * EE + nt * 16 + part * 8) = d;
    }
}

// ---------------------------------------------------------------------------
// K5: out-projection GEMM (double-buffered) + residual add, fp32 output.
//     grid=(M/128, E/128).
// ---------------------------------------------------------------------------
__global__ __launch_bounds__(256) void outproj_kernel(
    const unsigned short* __restrict__ hb, const unsigned short* __restrict__ wo,
    const float* __restrict__ x, float* __restrict__ out) {
    __shared__ __align__(16) unsigned short ldsA[2 * 128 * 40];
    __shared__ __align__(16) unsigned short ldsB[2 * 128 * 40];
    __shared__ __align__(16) float tbuf[8][16 * 20];
    int tid = threadIdx.x, wave = tid >> 5, lane = tid & 31;
    int m0 = blockIdx.x * 128, n0 = blockIdx.y * 128;

    int srow = tid >> 1, shf = tid & 1;
    const unsigned short* Ap = hb + (size_t)(m0 + srow) * EE + shf * 16;
    const unsigned short* Bp = wo + (size_t)(n0 + srow) * EE + shf * 16;
    unsigned short* sa = ldsA + srow * 40 + shf * 16;
    unsigned short* sb = ldsB + srow * 40 + shf * 16;

    v8f acc[8] = {};
#ifdef HAVE_ASYNC_LDS
    ASYNC_CP16(Ap, sa); ASYNC_CP16(Ap + 8, sa + 8);
    ASYNC_CP16(Bp, sb); ASYNC_CP16(Bp + 8, sb + 8);
#else
    {
        uint4 ra0 = *(const uint4*)Ap, ra1 = *(const uint4*)(Ap + 8);
        uint4 rb0 = *(const uint4*)Bp, rb1 = *(const uint4*)(Bp + 8);
        *(uint4*)sa = ra0; *(uint4*)(sa + 8) = ra1;
        *(uint4*)sb = rb0; *(uint4*)(sb + 8) = rb1;
    }
#endif
    int cur = 0;
    for (int k0 = 0; k0 < EE; k0 += 32) {
        ASYNC_FENCE();
        __syncthreads();
        int nxt = cur ^ 1;
        bool more = (k0 + 32 < EE);
#ifdef HAVE_ASYNC_LDS
        if (more) {
            ASYNC_CP16(Ap + k0 + 32, sa + nxt * 5120);
            ASYNC_CP16(Ap + k0 + 40, sa + nxt * 5120 + 8);
            ASYNC_CP16(Bp + k0 + 32, sb + nxt * 5120);
            ASYNC_CP16(Bp + k0 + 40, sb + nxt * 5120 + 8);
        }
#else
        uint4 na0 = {}, na1 = {}, nb0 = {}, nb1 = {};
        if (more) {
            na0 = *(const uint4*)(Ap + k0 + 32); na1 = *(const uint4*)(Ap + k0 + 40);
            nb0 = *(const uint4*)(Bp + k0 + 32); nb1 = *(const uint4*)(Bp + k0 + 40);
        }
#endif
        const unsigned short* ar =
            ldsA + cur * 5120 + (wave * 16 + (lane & 15)) * 40 + ((lane >> 4) << 4);
        bf16x16 af = load_frag(ar, ar + 8);
        const unsigned short* bbase =
            ldsB + cur * 5120 + (lane & 15) * 40 + ((lane >> 4) << 4);
        bf16x16 bfr[8];
        #pragma unroll
        for (int nt = 0; nt < 8; nt++)
            bfr[nt] = load_frag(bbase + nt * 640, bbase + nt * 640 + 8);
        #pragma unroll
        for (int nt = 0; nt < 8; nt++)
            acc[nt] = wmma_bf16(af, bfr[nt], acc[nt]);
#ifndef HAVE_ASYNC_LDS
        if (more) {
            unsigned short* da = sa + nxt * 5120;
            unsigned short* db = sb + nxt * 5120;
            *(uint4*)da = na0; *(uint4*)(da + 8) = na1;
            *(uint4*)db = nb0; *(uint4*)(db + 8) = nb1;
        }
#endif
        cur = nxt;
    }
    float* tb = tbuf[wave];
    size_t rowbase = (size_t)(m0 + wave * 16);
    #pragma unroll
    for (int nt = 0; nt < 8; nt++) {
        #pragma unroll
        for (int r = 0; r < 8; r++) {
            int row = (lane < 16) ? r : 8 + r;
            tb[row * 20 + (lane & 15)] = acc[nt][r];
        }
        asm volatile("s_wait_dscnt 0x0" ::: "memory");
        int row = lane >> 1, part = lane & 1;
        const float* src = tb + row * 20 + part * 8;
        float4 v0 = ((const float4*)src)[0];
        float4 v1 = ((const float4*)src)[1];
        size_t gi = (rowbase + row) * EE + n0 + nt * 16 + part * 8;
        float4 x0 = *(const float4*)(x + gi);
        float4 x1 = *(const float4*)(x + gi + 4);
        v0.x += x0.x; v0.y += x0.y; v0.z += x0.z; v0.w += x0.w;
        v1.x += x1.x; v1.y += x1.y; v1.z += x1.z; v1.w += x1.w;
        *(float4*)(out + gi) = v0;
        *(float4*)(out + gi + 4) = v1;
    }
}

// ---------------------------------------------------------------------------
extern "C" void kernel_launch(void* const* d_in, const int* in_sizes, int n_in,
                              void* d_out, int out_size, void* d_ws, size_t ws_size,
                              hipStream_t stream) {
    const float* x    = (const float*)d_in[0];
    const float* cosb = (const float*)d_in[1];
    const float* sinb = (const float*)d_in[2];
    const float* wqkv = (const float*)d_in[3];
    const float* wsw  = (const float*)d_in[4];
    const float* wout = (const float*)d_in[5];

    char* ws = (char*)d_ws;
    size_t off = 0;
    const size_t NTOK = (size_t)BB * TT;           // 8192
    unsigned short* xn  = (unsigned short*)(ws + off); off += NTOK * EE * 2;            // 32MB
    unsigned short* qb  = (unsigned short*)(ws + off); off += (size_t)BB*HH*TT*DD * 2;  // 32MB
    unsigned short* kb  = (unsigned short*)(ws + off); off += (size_t)BB*HH*TT*DD * 2;  // 32MB
    unsigned short* vT  = (unsigned short*)(ws + off); off += (size_t)BB*HH*DD*TT * 2;  // 32MB
    unsigned short* yb  = (unsigned short*)(ws + off); off += NTOK * EE * 2;            // 32MB
    unsigned short* hbuf= (unsigned short*)(ws + off); off += NTOK * EE * 2;            // 32MB
    unsigned short* wqkv_bf = (unsigned short*)(ws + off); off += (size_t)3*HH*DD*EE*2; // 24MB
    unsigned short* wsw_bf  = (unsigned short*)(ws + off); off += (size_t)2*EE*EE*2;    // 16MB
    unsigned short* wo_bf   = (unsigned short*)(ws + off); off += (size_t)EE*EE*2;      //  8MB

    int n1 = 3 * HH * DD * EE;
    int n2 = 2 * EE * EE;
    int n3 = EE * EE;
    cvt_f32_bf16<<<n1 / 1024, 256, 0, stream>>>(wqkv, wqkv_bf, n1);
    cvt_f32_bf16<<<n2 / 1024, 256, 0, stream>>>(wsw,  wsw_bf,  n2);
    cvt_f32_bf16<<<n3 / 1024, 256, 0, stream>>>(wout, wo_bf,   n3);

    rmsnorm_cast<<<(int)NTOK, 256, 0, stream>>>(x, xn);

    qkv_rope_kernel<<<dim3((int)NTOK / 128, 3 * HH), 256, 0, stream>>>(
        xn, wqkv_bf, cosb, sinb, qb, kb, vT);

    attn_kernel<<<dim3(TT / 128, BB * HH), 256, 0, stream>>>(qb, kb, vT, yb);

    swiglu_kernel<<<dim3((int)NTOK / 128, EE / 64), 256, 0, stream>>>(yb, wsw_bf, hbuf);

    outproj_kernel<<<dim3((int)NTOK / 128, EE / 128), 256, 0, stream>>>(
        hbuf, wo_bf, x, (float*)d_out);
}
